// LSTM_13383118094551
// MI455X (gfx1250) — compile-verified, matrix operand order checked by reading, not run on hardware
//
#include <hip/hip_runtime.h>

// ============================================================================
// LSTM for MI455X (gfx1250, wave32, WMMA bf16 16x16x32, f32 accumulate)
//
// Phase 1 (parallel):  gates_x[T*B, 4H] = bf16(X)[T*B, D] @ bf16(Wx)[D,4H] + bias
// Phase 2 (sequential, persistent 16-WG grid + atomic grid barrier):
//     per step t: gates = gates_x[t] + bf16(h_t) @ bf16(Wh);  elementwise LSTM
//     W_h slice LDS-resident (staged once); h_t staged to LDS each step, so the
//     serial critical path runs on ds_load_b128 + v_wmma only.
//
// Workspace layout (bytes), total ~306 MB:
//   Xbf      : T*B*D        bf16  =  33,554,432
//   Wx_pack  : D*4H         bf16  =   2,097,152   (WMMA-B swizzled tiles)
//   Wh_pack  : H*4H         bf16  =   2,097,152   (WMMA-B swizzled tiles)
//   h_buf    : 2*B*H        bf16  =      65,536   (double buffered)
//   counter  : 64B
//   gates_x  : T*B*4H       f32   = 268,435,456
// ============================================================================

typedef __attribute__((ext_vector_type(16))) __bf16 v16bf;
typedef __attribute__((ext_vector_type(8)))  __bf16 v8bf;
typedef __attribute__((ext_vector_type(8)))  float  v8f;

#define T_STEPS 1024
#define BATCH   32
#define DIM     512
#define HID     512
#define GATES   2048                 // 4*HID
#define ROWS    (T_STEPS * BATCH)    // 32768
#define KT512   (DIM / 32)           // 16 k-steps of 32
#define NWG     16                   // recurrent workgroups (persistent)

// dynamic LDS carve-up for the recurrent kernel
#define WH_LDS_BYTES  (8 * 16 * 32 * 16 * 2)   // 131072: 8 n-tiles x 16 k-tiles, packed
#define H_LDS_BYTES   (BATCH * HID * 2)        //  32768: h_t, bf16 row-major
#define G_LDS_BYTES   (BATCH * 128 * 4)        //  16384: gate slice f|i|o|g x 32 cols
#define C_LDS_BYTES   (BATCH * 32 * 4)         //   4096: cell state slice
#define REC_LDS_TOTAL (WH_LDS_BYTES + H_LDS_BYTES + G_LDS_BYTES + C_LDS_BYTES) // 184320

__device__ __forceinline__ unsigned short f32_to_bf16(float f) {
    unsigned u = __float_as_uint(f);
    u += 0x7FFFu + ((u >> 16) & 1u);          // round-to-nearest-even
    return (unsigned short)(u >> 16);
}

__device__ __forceinline__ float sigm(float x) { return 1.0f / (1.0f + __expf(-x)); }

// A-matrix (16x32 bf16) tile load from a row-major bf16 row.
// Per ISA 7.12.2 (16-bit A layout): lane l holds M=l%16; lanes 0-15/16-31
// hold K halves; each lane's 16 values = two contiguous 8-element chunks.
__device__ __forceinline__ v16bf load_a16x32(const unsigned short* row, int kbase, int half) {
    v8bf lo = *reinterpret_cast<const v8bf*>(row + kbase + half * 8);
    v8bf hi = *reinterpret_cast<const v8bf*>(row + kbase + 16 + half * 8);
    v16bf a;
#pragma unroll
    for (int i = 0; i < 8; ++i) { a[i] = lo[i]; a[i + 8] = hi[i]; }
    return a;
}

__device__ __forceinline__ v8f wmma_bf16(v16bf a, v16bf b, v8f c) {
    return __builtin_amdgcn_wmma_f32_16x16x32_bf16(false, a, false, b, (short)0, c, false, false);
}

// ---------------------------------------------------------------- conversions
__global__ void conv_x_kernel(const float* __restrict__ X, unsigned short* __restrict__ Xbf) {
    size_t id = (size_t)blockIdx.x * blockDim.x + threadIdx.x;
    if (id < (size_t)ROWS * DIM) Xbf[id] = f32_to_bf16(X[id]);
}

// Swizzle W (fp32 [D+H, 4H]) into WMMA-B tiles, bf16.
// packed[(ntile*16 + kt)*32*16 + lane*16 + s] = W[kbase + k(lane,s)][ntile*16 + lane%16]
// with k(l,s) = kt*32 + (s<8 ? half*8 + s : 16 + half*8 + (s-8)), half = l/16.
__global__ void pack_w_kernel(const float* __restrict__ W,
                              unsigned short* __restrict__ Wxp,
                              unsigned short* __restrict__ Whp) {
    size_t id = (size_t)blockIdx.x * blockDim.x + threadIdx.x;   // 0 .. 2*1048576
    if (id >= 2u * 1048576u) return;
    int which = (id >= 1048576u);
    size_t e = id & 1048575u;
    int s     = (int)(e & 15u);
    int lane  = (int)((e >> 4) & 31u);
    int kt    = (int)((e >> 9) & 15u);
    int ntile = (int)(e >> 13);
    int half  = lane >> 4;
    int kk    = (s < 8) ? (half * 8 + s) : (16 + half * 8 + (s - 8));
    int k     = kt * 32 + kk;
    int n     = ntile * 16 + (lane & 15);
    int row   = which ? (DIM + k) : k;                 // Wx rows [0,512), Wh rows [512,1024)
    unsigned short v = f32_to_bf16(W[(size_t)row * GATES + n]);
    (which ? Whp : Wxp)[e] = v;
}

// out[0] = broadcast(init_hidden); h_buf[0] = bf16 of it; reset grid barrier.
__global__ void init_kernel(const float* __restrict__ h0,
                            float* __restrict__ out,
                            unsigned short* __restrict__ h_buf,
                            unsigned int* __restrict__ counter) {
    int id = blockIdx.x * blockDim.x + threadIdx.x;    // 0 .. B*H
    if (id == 0) *counter = 0u;
    if (id < BATCH * HID) {
        float v = h0[id % HID];
        out[id]   = v;                                  // row t=0 of output
        h_buf[id] = f32_to_bf16(v);                     // parity-0 buffer
    }
}

// ------------------------------------------------- phase 1: gates_x GEMM
// grid: (ROWS/32, GATES/64). 256 thr = 8 waves = 2 Mtiles x 4 Ntiles.
__global__ void gemm_x_kernel(const unsigned short* __restrict__ Xbf,
                              const unsigned short* __restrict__ Wxp,
                              const float* __restrict__ bias,
                              float* __restrict__ gates_x) {
    int wave = threadIdx.x >> 5;
    int lane = threadIdx.x & 31;
    int half = lane >> 4;
    int mtile = blockIdx.x * 2 + (wave >> 2);
    int ntile = blockIdx.y * 4 + (wave & 3);
    int mrow  = mtile * 16 + (lane & 15);

    const unsigned short* arow = Xbf + (size_t)mrow * DIM;
    v8f acc = {};
#pragma unroll
    for (int kt = 0; kt < KT512; ++kt) {
        v16bf a = load_a16x32(arow, kt * 32, half);
        v16bf b = *reinterpret_cast<const v16bf*>(
            Wxp + (((size_t)(ntile * 16 + kt)) * 32 + lane) * 16);
        acc = wmma_bf16(a, b, acc);
    }
    int n = ntile * 16 + (lane & 15);
    float bv = bias[n];
    int mbase = mtile * 16 + half * 8;                 // C/D layout: lane l -> N=l%16, M=(l/16)*8+r
#pragma unroll
    for (int r = 0; r < 8; ++r)
        gates_x[(size_t)(mbase + r) * GATES + n] = acc[r] + bv;
}

// ------------------------------------------------- phase 2: persistent recurrence
// 16 WGs x 256 threads. WG j owns H-columns [32j, 32j+32) -> gate cols
// {g*512 + 32j .. +32} for g=f,i,o,g. Wave w: gate g=w/2, ntile = g*32 + 2j + (w&1).
__global__ void lstm_rec_kernel(const unsigned short* __restrict__ Whp,
                                const float* __restrict__ gates_x,
                                const float* __restrict__ c0,
                                unsigned short* __restrict__ h_buf,
                                float* __restrict__ out,
                                unsigned int* __restrict__ counter) {
    extern __shared__ char smem[];
    unsigned short* wh_lds = (unsigned short*)(smem);                       // packed B tiles
    unsigned short* h_lds  = (unsigned short*)(smem + WH_LDS_BYTES);        // h_t row-major
    float*          g_lds  = (float*)(smem + WH_LDS_BYTES + H_LDS_BYTES);   // [32][128]
    float*          c_lds  = (float*)(smem + WH_LDS_BYTES + H_LDS_BYTES + G_LDS_BYTES);

    const int j    = blockIdx.x;
    const int wave = threadIdx.x >> 5;
    const int lane = threadIdx.x & 31;
    const int half = lane >> 4;

    // init cell state slice
    for (int idx = threadIdx.x; idx < BATCH * 32; idx += blockDim.x)
        c_lds[idx] = c0[j * 32 + (idx & 31)];

    // Stage this WG's Wh slice (8 n-tiles x 16 k-tiles, packed) into LDS once.
    // One packed tile = 32 lanes x 16 ushort = 1KB = 64 uint4 chunks.
    {
        const uint4* src = (const uint4*)Whp;
        uint4*       dst = (uint4*)wh_lds;
        for (int fi = threadIdx.x; fi < 8 * 16 * 64; fi += blockDim.x) {
            int tile = fi >> 6;                 // local tile = w*16 + kt
            int rem  = fi & 63;
            int w    = tile >> 4;
            int kt   = tile & 15;
            int g    = w >> 1;
            int ntg  = g * 32 + j * 2 + (w & 1);
            dst[fi] = src[(size_t)(ntg * 16 + kt) * 64 + rem];
        }
    }

    const int gblk = wave >> 1;                         // 0..3 = f,i,o,g
    const int ntg  = gblk * 32 + j * 2 + (wave & 1);    // global gate n-tile
    const int ncol = ntg * 16 + (lane & 15);
    const int sc   = wave * 16 + (lane & 15);           // column in g_lds slice
    __syncthreads();

    for (int t = 0; t < T_STEPS; ++t) {
        const float* gx = gates_x + (size_t)t * BATCH * GATES;

        // stage h_t (32 KB bf16) into LDS once per step: 2048 uint4 / 256 thr
        {
            const uint4* src = (const uint4*)(h_buf + (size_t)(t & 1) * BATCH * HID);
            uint4*       dst = (uint4*)h_lds;
            for (int fi = threadIdx.x; fi < (BATCH * HID * 2) / 16; fi += blockDim.x)
                dst[fi] = src[fi];
        }

        if (t + 1 < T_STEPS)   // pull next step's gates_x block toward L2/WGP$
            __builtin_prefetch(gates_x + (size_t)(t + 1) * BATCH * GATES
                                       + (size_t)threadIdx.x * 256, 0, 0);

        // accumulators start from precomputed x-gates (+bias)
        v8f acc0, acc1;
#pragma unroll
        for (int r = 0; r < 8; ++r) {
            acc0[r] = gx[(size_t)(half * 8 + r) * GATES + ncol];
            acc1[r] = gx[(size_t)(16 + half * 8 + r) * GATES + ncol];
        }
        __syncthreads();                                // h_lds ready

        const unsigned short* a0row = h_lds + (size_t)(lane & 15) * HID;
        const unsigned short* a1row = h_lds + (size_t)(16 + (lane & 15)) * HID;
        const unsigned short* btile = wh_lds + (size_t)wave * 16 * 512;   // wave's 16 k-tiles
#pragma unroll
        for (int kt = 0; kt < KT512; ++kt) {
            v16bf b  = *reinterpret_cast<const v16bf*>(btile + ((size_t)kt * 32 + lane) * 16);
            v16bf a0 = load_a16x32(a0row, kt * 32, half);
            v16bf a1 = load_a16x32(a1row, kt * 32, half);
            acc0 = wmma_bf16(a0, b, acc0);
            acc1 = wmma_bf16(a1, b, acc1);
        }
#pragma unroll
        for (int r = 0; r < 8; ++r) {
            g_lds[(half * 8 + r) * 128 + sc]      = acc0[r];
            g_lds[(16 + half * 8 + r) * 128 + sc] = acc1[r];
        }
        __syncthreads();

        // elementwise LSTM cell update for this WG's [32 x 32] slice
        unsigned short* hbn = h_buf + (size_t)((t + 1) & 1) * BATCH * HID;
        for (int idx = threadIdx.x; idx < BATCH * 32; idx += blockDim.x) {
            int m = idx >> 5, cc = idx & 31;
            float fg = sigm(g_lds[m * 128 + cc]);
            float ig = sigm(g_lds[m * 128 + 32 + cc]);
            float og = sigm(g_lds[m * 128 + 64 + cc]);
            float gg = tanhf(g_lds[m * 128 + 96 + cc]);
            float cn = fg * c_lds[idx] + ig * gg;
            c_lds[idx] = cn;
            float hn = og * tanhf(cn);
            int col = j * 32 + cc;
            out[(size_t)(t + 1) * BATCH * HID + (size_t)m * HID + col] = hn;
            hbn[(size_t)m * HID + col] = f32_to_bf16(hn);
        }
        __threadfence();                                // release h_buf writes (device scope)
        __syncthreads();
        if (threadIdx.x == 0) {                         // grid-wide barrier (monotonic counter)
            atomicAdd(counter, 1u);
            unsigned target = (unsigned)(t + 1) * gridDim.x;
            while (__hip_atomic_load(counter, __ATOMIC_ACQUIRE,
                                     __HIP_MEMORY_SCOPE_AGENT) < target)
                __builtin_amdgcn_s_sleep(1);
        }
        __syncthreads();
        __threadfence();                                // acquire: invalidate stale WGP$ lines
    }
}

// ---------------------------------------------------------------- launcher
extern "C" void kernel_launch(void* const* d_in, const int* in_sizes, int n_in,
                              void* d_out, int out_size, void* d_ws, size_t ws_size,
                              hipStream_t stream) {
    const float* X    = (const float*)d_in[0];   // [T,B,D]
    const float* W    = (const float*)d_in[1];   // [D+H, 4H]
    const float* bias = (const float*)d_in[2];   // [1, 4H]
    const float* h0   = (const float*)d_in[3];   // [1, H]
    const float* c0   = (const float*)d_in[4];   // [1, H]
    float* out        = (float*)d_out;           // [T+1, B, H]

    char* ws = (char*)d_ws;
    unsigned short* Xbf   = (unsigned short*)(ws);
    unsigned short* Wxp   = (unsigned short*)(ws + (size_t)33554432);
    unsigned short* Whp   = (unsigned short*)(ws + (size_t)35651584);
    unsigned short* h_buf = (unsigned short*)(ws + (size_t)37748736);
    unsigned int*   cnt   = (unsigned int*)  (ws + (size_t)37814272);
    float*          gx    = (float*)         (ws + (size_t)37814336);
    (void)in_sizes; (void)n_in; (void)out_size; (void)ws_size; // needs ~306 MB ws

    // opt in to >64KB dynamic LDS for the persistent recurrent kernel
    hipFuncSetAttribute((const void*)lstm_rec_kernel,
                        hipFuncAttributeMaxDynamicSharedMemorySize, REC_LDS_TOTAL);

    conv_x_kernel<<<(ROWS * DIM + 255) / 256, 256, 0, stream>>>(X, Xbf);
    pack_w_kernel<<<(2 * 1048576 + 255) / 256, 256, 0, stream>>>(W, Wxp, Whp);
    init_kernel<<<(BATCH * HID + 255) / 256, 256, 0, stream>>>(h0, out, h_buf, cnt);
    dim3 gg(ROWS / 32, GATES / 64, 1);
    gemm_x_kernel<<<gg, 256, 0, stream>>>(Xbf, Wxp, bias, gx);
    lstm_rec_kernel<<<NWG, 256, REC_LDS_TOTAL, stream>>>(Whp, gx, c0, h_buf, out, cnt);
}
